// Toy_Transformer_58480274703173
// MI455X (gfx1250) — matrix operand order repaired and checked by reading, not compiled, not run on hardware
//
#include <hip/hip_runtime.h>
#include <hip/hip_bf16.h>
#include <math.h>

// ---------------------------------------------------------------------------
// Toy transformer block for MI455X (gfx1250, wave32, WMMA).
// B=2, L=2048, D=1024, H=16, HD=64.  All GEMMs + attention use
// v_wmma_f32_16x16x32_bf16 with f32 accumulation; GEMM staging uses the
// gfx1250 async global->LDS path (ASYNCcnt) with double buffering.
// ---------------------------------------------------------------------------

#define B_  2
#define L_  2048
#define D_  1024
#define H_  16
#define HD_ 64
#define M_  (B_ * L_)          // 4096 rows of activations
#define EPSF 1e-5f

typedef __bf16 bf16;
typedef __attribute__((ext_vector_type(16))) __bf16 v16bf;
typedef __attribute__((ext_vector_type(8)))  float  v8f;

union Frag16 { v16bf v; uint4 u[2]; };

__device__ __forceinline__ v8f wmma_bf16(v16bf a, v16bf b, v8f c) {
    // (neg_a, A, neg_b, B, c_mod, C, reuse_a, reuse_b)
    return __builtin_amdgcn_wmma_f32_16x16x32_bf16(false, a, false, b,
                                                   (short)0, c, false, false);
}

// ---- gfx1250 async global->LDS copy (ASYNCcnt) ----------------------------
#if __has_builtin(__builtin_amdgcn_global_load_async_to_lds_b128) && \
    __has_builtin(__builtin_amdgcn_s_wait_asynccnt)
#define HAS_ASYNC_LDS 1
#else
#define HAS_ASYNC_LDS 0
#endif

// builtin parameter types (from clang diagnostic): AS1 / AS3 pointers to v4i
typedef __attribute__((__vector_size__(4 * sizeof(int)))) int vint4_t;
typedef __attribute__((address_space(1))) vint4_t* gptr_v4i;
typedef __attribute__((address_space(3))) vint4_t* sptr_v4i;

__device__ __forceinline__ void copy16_g2lds(const bf16* g, bf16* l) {
#if HAS_ASYNC_LDS
    // generic->AS pointer conversion: low 32 bits of a generic shared address
    // are the LDS offset; global generic == AS1 numerically.
    __builtin_amdgcn_global_load_async_to_lds_b128(
        (gptr_v4i)(uintptr_t)g,
        (sptr_v4i)(uintptr_t)l,
        0, 0);
#else
    *(uint4*)l = *(const uint4*)g;
#endif
}

__device__ __forceinline__ void wait_async_le3() {
#if HAS_ASYNC_LDS
    __builtin_amdgcn_s_wait_asynccnt(3);
#endif
}
__device__ __forceinline__ void wait_async_le0() {
#if HAS_ASYNC_LDS
    __builtin_amdgcn_s_wait_asynccnt(0);
#endif
}

// ---------------------------------------------------------------------------
// Weight transpose + f32 -> bf16 convert:  in[K][N] f32  ->  out[N][K] bf16
// grid: (N/32, K/32), block: 256
// ---------------------------------------------------------------------------
__global__ __launch_bounds__(256)
void tconv_kernel(const float* __restrict__ in, bf16* __restrict__ outT,
                  int K, int N) {
    __shared__ float tile[32][33];
    const int nb = blockIdx.x * 32;
    const int kb = blockIdx.y * 32;
    const int tx = threadIdx.x & 31;
    const int ty = threadIdx.x >> 5;   // 0..7
#pragma unroll
    for (int i = 0; i < 4; ++i) {
        int k = kb + ty + i * 8;
        tile[ty + i * 8][tx] = in[(size_t)k * N + nb + tx];
    }
    __syncthreads();
#pragma unroll
    for (int i = 0; i < 4; ++i) {
        int n = nb + ty + i * 8;
        outT[(size_t)n * K + kb + tx] = (bf16)tile[tx][ty + i * 8];
    }
}

// ---------------------------------------------------------------------------
// LayerNorm (torch semantics: unbiased std, eps added to std) -> bf16
// one block (256 threads) per row of 1024
// ---------------------------------------------------------------------------
__global__ __launch_bounds__(256)
void ln_kernel(const float* __restrict__ x, const float* __restrict__ g,
               const float* __restrict__ b, bf16* __restrict__ out) {
    const int row = blockIdx.x;
    const int t   = threadIdx.x;
    const float* xr = x + (size_t)row * D_;
    float v[4], s = 0.f, sq = 0.f;
#pragma unroll
    for (int i = 0; i < 4; ++i) {
        v[i] = xr[t + i * 256];
        s  += v[i];
        sq += v[i] * v[i];
    }
#pragma unroll
    for (int off = 16; off >= 1; off >>= 1) {
        s  += __shfl_xor(s,  off, 32);
        sq += __shfl_xor(sq, off, 32);
    }
    __shared__ float reds[8], redq[8];
    const int wave = t >> 5, lane = t & 31;
    if (lane == 0) { reds[wave] = s; redq[wave] = sq; }
    __syncthreads();
    float ts = 0.f, tq = 0.f;
#pragma unroll
    for (int w = 0; w < 8; ++w) { ts += reds[w]; tq += redq[w]; }
    const float mean = ts * (1.0f / D_);
    float var = (tq - (float)D_ * mean * mean) * (1.0f / (D_ - 1)); // ddof=1
    var = fmaxf(var, 0.0f);
    const float inv = 1.0f / (sqrtf(var) + EPSF);
#pragma unroll
    for (int i = 0; i < 4; ++i) {
        int c = t + i * 256;
        out[(size_t)row * D_ + c] = (bf16)((v[i] - mean) * inv * g[c] + b[c]);
    }
}

// ---------------------------------------------------------------------------
// WMMA GEMM:  C[M,N] = A[M,K](bf16 row-major) * Wt[N,K](bf16, pre-transposed)
// BM=128 BN=64 BK=32, 256 threads = 8 waves (4x2), each wave 32x32 output.
// Double-buffered async global->LDS staging (3 async ops/wave/stage).
// MODE 0: QKV scatter (q scaled 1/8; V stored transposed [HD][L])
// MODE 1: + bias + residual -> f32
// MODE 2: + bias, tanh-GELU -> bf16
// MODE 3: + bias + residual -> f32 (final output)
// ---------------------------------------------------------------------------
#define LDSTRIDE 40   // 32 bf16 + 8 pad; keeps 16B alignment per row

template <int MODE>
__global__ __launch_bounds__(256)
void gemm_bf16_kernel(const bf16* __restrict__ A, const bf16* __restrict__ Wt,
                      const float* __restrict__ bias, const float* __restrict__ res,
                      float* __restrict__ outF, bf16* __restrict__ outB,
                      bf16* __restrict__ qb, bf16* __restrict__ kb,
                      bf16* __restrict__ vt,
                      int M, int N, int K) {
    __shared__ bf16 As[2][128 * LDSTRIDE];
    __shared__ bf16 Bs[2][64 * LDSTRIDE];

    const int t     = threadIdx.x;
    const int lane  = t & 31;
    const int wave  = t >> 5;
    const int wm    = wave & 3;     // 0..3  (rows)
    const int wn    = wave >> 2;    // 0..1  (cols)
    const int lrow  = lane & 15;
    const int lhalf = lane >> 4;

    const int blockM = blockIdx.x * 128;
    const int blockN = blockIdx.y * 64;

    // per-thread staging coordinates (16B each)
    const int arow0 = t >> 2;             // A rows t/4 and t/4+64
    const int ac8   = (t & 3) * 8;
    const int brow  = t >> 2;             // B row
    const int bc8   = (t & 3) * 8;

    auto stage = [&](int k0, int buf) {
        copy16_g2lds(A  + (size_t)(blockM + arow0)      * K + k0 + ac8,
                     &As[buf][arow0 * LDSTRIDE + ac8]);
        copy16_g2lds(A  + (size_t)(blockM + arow0 + 64) * K + k0 + ac8,
                     &As[buf][(arow0 + 64) * LDSTRIDE + ac8]);
        copy16_g2lds(Wt + (size_t)(blockN + brow)       * K + k0 + bc8,
                     &Bs[buf][brow * LDSTRIDE + bc8]);
    };

    v8f acc[2][2] = {};
    const int nk = K >> 5;
    int cur = 0;
    stage(0, 0);

    for (int kt = 0; kt < nk; ++kt) {
        const bool more = (kt + 1 < nk);
        if (more) stage((kt + 1) << 5, cur ^ 1);   // prefetch next tile (async)
        if (more) wait_async_le3(); else wait_async_le0();
        __syncthreads();

        // ---- fragments per ISA layouts ---------------------------------
        Frag16 af[2], bf_[2];
#pragma unroll
        for (int i = 0; i < 2; ++i) {
            const bf16* p = &As[cur][(wm * 32 + i * 16 + lrow) * LDSTRIDE];
            af[i].u[0] = *(const uint4*)(p + lhalf * 8);          // K 0..7 / 8..15
            af[i].u[1] = *(const uint4*)(p + 16 + lhalf * 8);     // K 16..23 / 24..31
        }
#pragma unroll
        for (int j = 0; j < 2; ++j) {
            const bf16* p = &Bs[cur][(wn * 32 + j * 16 + lrow) * LDSTRIDE];
            bf_[j].u[0] = *(const uint4*)(p + lhalf * 8);         // K 0..7 / 8..15
            bf_[j].u[1] = *(const uint4*)(p + 16 + lhalf * 8);    // K 16..23 / 24..31
        }
#pragma unroll
        for (int i = 0; i < 2; ++i)
#pragma unroll
            for (int j = 0; j < 2; ++j)
                acc[i][j] = wmma_bf16(af[i].v, bf_[j].v, acc[i][j]);

        __syncthreads();   // all readers done before buf `cur` is re-staged
        cur ^= 1;
    }

    // ---- epilogue ------------------------------------------------------
#pragma unroll
    for (int i = 0; i < 2; ++i)
#pragma unroll
        for (int j = 0; j < 2; ++j)
#pragma unroll
            for (int r = 0; r < 8; ++r) {
                const int row = blockM + wm * 32 + i * 16 + lhalf * 8 + r;
                const int col = blockN + wn * 32 + j * 16 + lrow;
                float val = acc[i][j][r];
                if constexpr (MODE == 0) {
                    const int three = col >> 10;
                    const int rem   = col & 1023;
                    const int hh    = rem >> 6;
                    const int hd    = rem & 63;
                    const int bb    = row >> 11;
                    const int lp    = row & 2047;
                    const size_t bh = (size_t)(bb * H_ + hh);
                    if (three == 0)
                        qb[(bh * L_ + lp) * HD_ + hd] = (bf16)(val * 0.125f);
                    else if (three == 1)
                        kb[(bh * L_ + lp) * HD_ + hd] = (bf16)val;
                    else
                        vt[(bh * HD_ + hd) * L_ + lp] = (bf16)val;   // transposed
                } else if constexpr (MODE == 1) {
                    val += bias[col] + res[(size_t)row * N + col];
                    outF[(size_t)row * N + col] = val;
                } else if constexpr (MODE == 2) {
                    val += bias[col];
                    const float x3 = val * val * val;
                    const float gl = 0.5f * val *
                        (1.0f + tanhf(0.7978845608028654f * (val + 0.044715f * x3)));
                    outB[(size_t)row * N + col] = (bf16)gl;
                } else {
                    val += bias[col] + res[(size_t)row * N + col];
                    outF[(size_t)row * N + col] = val;
                }
            }
}

// ---------------------------------------------------------------------------
// Flash attention, fully WMMA.  block = 128 threads = 4 waves; each wave owns
// one 16-row Q tile.  grid = (B*H, L/64).
// qb/kb: [B,H,L,HD] bf16 (q pre-scaled by 1/sqrt(HD)); vt: [B,H,HD,L] bf16.
// ctx out: [B,L,H*HD] bf16 (ready as next GEMM's A matrix).
// ---------------------------------------------------------------------------
__global__ __launch_bounds__(128)
void attn_kernel(const bf16* __restrict__ qb, const bf16* __restrict__ kb,
                 const bf16* __restrict__ vt, bf16* __restrict__ ctx) {
    const int bh    = blockIdx.x;            // 0..31
    const int b     = bh >> 4;
    const int h     = bh & 15;
    const int t     = threadIdx.x;
    const int wave  = t >> 5;
    const int lane  = t & 31;
    const int lrow  = lane & 15;
    const int lhalf = lane >> 4;
    const int qbase = blockIdx.y * 64 + wave * 16;

    const bf16* Q  = qb + (size_t)bh * L_ * HD_;
    const bf16* Kp = kb + (size_t)bh * L_ * HD_;
    const bf16* Vt = vt + (size_t)bh * HD_ * L_;

    __shared__ bf16 Pl[4][16 * LDSTRIDE];    // per-wave P staging
    bf16* Pw = &Pl[wave][0];

    // Q fragments (A-matrix 16x32), hd 0..31 and 32..63, loaded once
    Frag16 q0, q1;
    {
        const bf16* p = Q + (size_t)(qbase + lrow) * HD_;
        q0.u[0] = *(const uint4*)(p + lhalf * 16);
        q0.u[1] = *(const uint4*)(p + lhalf * 16 + 8);
        q1.u[0] = *(const uint4*)(p + 32 + lhalf * 16);
        q1.u[1] = *(const uint4*)(p + 32 + lhalf * 16 + 8);
    }

    v8f o[4] = {};
    float mrun[8], lrun[8];
#pragma unroll
    for (int r = 0; r < 8; ++r) { mrun[r] = -1e30f; lrun[r] = 0.f; }

    const int ktmax = (qbase + 15) >> 5;     // causal: only tiles on/below diag
    for (int kt = 0; kt <= ktmax; ++kt) {
        const int kbase = kt * 32;

        // prefetch next key/value tiles into cache (global_prefetch_b8)
        if (kt < ktmax) {
            __builtin_prefetch(Kp + (size_t)(kbase + 32 + lrow) * HD_, 0, 1);
            __builtin_prefetch(Vt + (size_t)lrow * L_ + kbase + 32, 0, 1);
        }

        // ---- S = Q * K^T  (two 16x16 tiles over 32 keys) ---------------
        v8f s0 = {}, s1 = {};
        {
            const bf16* kp0 = Kp + (size_t)(kbase + lrow) * HD_;        // j=0
            const bf16* kp1 = Kp + (size_t)(kbase + 16 + lrow) * HD_;   // j=1
            Frag16 kf;
            kf.u[0] = *(const uint4*)(kp0 + lhalf * 16);
            kf.u[1] = *(const uint4*)(kp0 + lhalf * 16 + 8);
            s0 = wmma_bf16(q0.v, kf.v, s0);
            kf.u[0] = *(const uint4*)(kp0 + 32 + lhalf * 16);
            kf.u[1] = *(const uint4*)(kp0 + 32 + lhalf * 16 + 8);
            s0 = wmma_bf16(q1.v, kf.v, s0);
            kf.u[0] = *(const uint4*)(kp1 + lhalf * 16);
            kf.u[1] = *(const uint4*)(kp1 + lhalf * 16 + 8);
            s1 = wmma_bf16(q0.v, kf.v, s1);
            kf.u[0] = *(const uint4*)(kp1 + 32 + lhalf * 16);
            kf.u[1] = *(const uint4*)(kp1 + 32 + lhalf * 16 + 8);
            s1 = wmma_bf16(q1.v, kf.v, s1);
        }

        // ---- causal mask + online softmax ------------------------------
        float sm0[8], sm1[8];
        const bool domask = (kbase + 31) > qbase;
#pragma unroll
        for (int r = 0; r < 8; ++r) {
            sm0[r] = s0[r];
            sm1[r] = s1[r];
            if (domask) {
                const int row = qbase + lhalf * 8 + r;
                const int c0  = kbase + lrow;
                if (c0 > row)      sm0[r] = -1e30f;
                if (c0 + 16 > row) sm1[r] = -1e30f;
            }
        }
#pragma unroll
        for (int r = 0; r < 8; ++r) {
            float mt = fmaxf(sm0[r], sm1[r]);
#pragma unroll
            for (int off = 8; off >= 1; off >>= 1)   // 16-lane half reduction
                mt = fmaxf(mt, __shfl_xor(mt, off, 32));
            const float mnew = fmaxf(mrun[r], mt);
            const float sc   = __expf(mrun[r] - mnew);
            const float p0   = __expf(sm0[r] - mnew);
            const float p1   = __expf(sm1[r] - mnew);
            float rs = p0 + p1;
#pragma unroll
            for (int off = 8; off >= 1; off >>= 1)
                rs += __shfl_xor(rs, off, 32);
            lrun[r] = lrun[r] * sc + rs;
            mrun[r] = mnew;
#pragma unroll
            for (int jj = 0; jj < 4; ++jj) o[jj][r] *= sc;
            // stage P (D-layout -> LDS) for re-load in A-layout
            Pw[(lhalf * 8 + r) * LDSTRIDE + lrow]      = (bf16)p0;
            Pw[(lhalf * 8 + r) * LDSTRIDE + 16 + lrow] = (bf16)p1;
        }
        asm volatile("s_wait_dscnt 0x0" ::: "memory");

        Frag16 pa;
        {
            const bf16* p = &Pw[lrow * LDSTRIDE];
            pa.u[0] = *(const uint4*)(p + lhalf * 16);
            pa.u[1] = *(const uint4*)(p + lhalf * 16 + 8);
        }

        // ---- O += P * V  (V transposed: contiguous B fragments) --------
#pragma unroll
        for (int j = 0; j < 4; ++j) {
            const bf16* vp = Vt + (size_t)(j * 16 + lrow) * L_ + kbase + lhalf * 16;
            Frag16 vf;
            vf.u[0] = *(const uint4*)(vp);
            vf.u[1] = *(const uint4*)(vp + 8);
            o[j] = wmma_bf16(pa.v, vf.v, o[j]);
        }
    }

    // ---- normalize and write ctx [B, L, H*HD] --------------------------
#pragma unroll
    for (int r = 0; r < 8; ++r) {
        const float inv = 1.0f / lrun[r];
        const int qpos  = qbase + lhalf * 8 + r;
        const size_t ro = ((size_t)b * L_ + qpos) * D_ + h * HD_;
#pragma unroll
        for (int j = 0; j < 4; ++j)
            ctx[ro + j * 16 + lrow] = (bf16)(o[j][r] * inv);
    }
}

// ---------------------------------------------------------------------------
// Host-side launch
// ---------------------------------------------------------------------------
extern "C" void kernel_launch(void* const* d_in, const int* in_sizes, int n_in,
                              void* d_out, int out_size, void* d_ws, size_t ws_size,
                              hipStream_t stream) {
    (void)in_sizes; (void)n_in; (void)out_size; (void)ws_size;

    const float* x      = (const float*)d_in[0];
    const float* qkv_w  = (const float*)d_in[1];
    const float* out_w  = (const float*)d_in[2];
    const float* out_b  = (const float*)d_in[3];
    const float* ffn_w1 = (const float*)d_in[4];
    const float* ffn_b1 = (const float*)d_in[5];
    const float* ffn_w2 = (const float*)d_in[6];
    const float* ffn_b2 = (const float*)d_in[7];
    const float* ln1_g  = (const float*)d_in[8];
    const float* ln1_b  = (const float*)d_in[9];
    const float* ln2_g  = (const float*)d_in[10];
    const float* ln2_b  = (const float*)d_in[11];
    float* outp = (float*)d_out;

    // workspace carve-up (bytes)
    char* ws = (char*)d_ws;
    size_t off = 0;
    auto alloc = [&](size_t bytes) { char* p = ws + off; off += (bytes + 255) & ~(size_t)255; return p; };
    bf16* Wqkv_t = (bf16*)alloc((size_t)3 * D_ * D_ * 2);       // [3072][1024]
    bf16* Wout_t = (bf16*)alloc((size_t)D_ * D_ * 2);           // [1024][1024]
    bf16* W1_t   = (bf16*)alloc((size_t)4 * D_ * D_ * 2);       // [4096][1024]
    bf16* W2_t   = (bf16*)alloc((size_t)4 * D_ * D_ * 2);       // [1024][4096]
    bf16* h1     = (bf16*)alloc((size_t)M_ * D_ * 2);
    bf16* qbuf   = (bf16*)alloc((size_t)B_ * H_ * L_ * HD_ * 2);
    bf16* kbuf   = (bf16*)alloc((size_t)B_ * H_ * L_ * HD_ * 2);
    bf16* vtbuf  = (bf16*)alloc((size_t)B_ * H_ * HD_ * L_ * 2);
    bf16* ctx    = (bf16*)alloc((size_t)M_ * D_ * 2);
    float* ybuf  = (float*)alloc((size_t)M_ * D_ * 4);
    bf16* h2     = (bf16*)alloc((size_t)M_ * D_ * 2);
    bf16* a1     = (bf16*)alloc((size_t)M_ * 4 * D_ * 2);

    // 1) weight convert+transpose
    tconv_kernel<<<dim3(3 * D_ / 32, D_ / 32), 256, 0, stream>>>(qkv_w,  Wqkv_t, D_,     3 * D_);
    tconv_kernel<<<dim3(D_ / 32,     D_ / 32), 256, 0, stream>>>(out_w,  Wout_t, D_,     D_);
    tconv_kernel<<<dim3(4 * D_ / 32, D_ / 32), 256, 0, stream>>>(ffn_w1, W1_t,   D_,     4 * D_);
    tconv_kernel<<<dim3(D_ / 32, 4 * D_ / 32), 256, 0, stream>>>(ffn_w2, W2_t,   4 * D_, D_);

    // 2) LN1 -> bf16
    ln_kernel<<<M_, 256, 0, stream>>>(x, ln1_g, ln1_b, h1);

    // 3) QKV projection (scatter epilogue)
    gemm_bf16_kernel<0><<<dim3(M_ / 128, (3 * D_) / 64), 256, 0, stream>>>(
        h1, Wqkv_t, nullptr, nullptr, nullptr, nullptr,
        qbuf, kbuf, vtbuf, M_, 3 * D_, D_);

    // 4) flash attention
    attn_kernel<<<dim3(B_ * H_, L_ / 64), 128, 0, stream>>>(qbuf, kbuf, vtbuf, ctx);

    // 5) out projection + bias + residual -> y (f32)
    gemm_bf16_kernel<1><<<dim3(M_ / 128, D_ / 64), 256, 0, stream>>>(
        ctx, Wout_t, out_b, x, ybuf, nullptr,
        nullptr, nullptr, nullptr, M_, D_, D_);

    // 6) LN2 -> bf16
    ln_kernel<<<M_, 256, 0, stream>>>(ybuf, ln2_g, ln2_b, h2);

    // 7) FFN1 + bias + GELU -> bf16
    gemm_bf16_kernel<2><<<dim3(M_ / 128, (4 * D_) / 64), 256, 0, stream>>>(
        h2, W1_t, ffn_b1, nullptr, nullptr, a1,
        nullptr, nullptr, nullptr, M_, 4 * D_, D_);

    // 8) FFN2 + bias + residual -> final output (f32)
    gemm_bf16_kernel<3><<<dim3(M_ / 128, D_ / 64), 256, 0, stream>>>(
        a1, W2_t, ffn_b2, ybuf, outp, nullptr,
        nullptr, nullptr, nullptr, M_, D_, 4 * D_);
}
